// FraudRGTAN_33964601377359
// MI455X (gfx1250) — matrix operand order repaired and checked by reading, not compiled, not run on hardware
//
#include <hip/hip_runtime.h>
#include <hip/hip_bf16.h>

// FraudRGTAN for MI455X (gfx1250): bf16 WMMA GEMMs (LDS-staged weights) +
// fused multi-node MHA + atomic segment-softmax graph conv.
// Layouts follow CDNA5 ISA VGPR tables. Compile-only target.

#define N_NODES 30000
#define N_EDGES 480000
#define N_LAYERS 2
#define N_HEADSC 4
#define D_HIDC 128
#define D_INC 128
#define D_NEIC 64
#define S_NEIC 10
#define MHA_WAVES 8

typedef __attribute__((ext_vector_type(16))) __bf16 v16bf;
typedef __attribute__((ext_vector_type(8)))  float  v8f;

static __device__ __forceinline__ __bf16 f2bf(float f) { return (__bf16)f; }

// ---- order-preserving float <-> uint for atomicMax-based segment max ----
static __device__ __forceinline__ unsigned fflip(float f) {
  unsigned u = __float_as_uint(f);
  return (u & 0x80000000u) ? ~u : (u | 0x80000000u);
}
static __device__ __forceinline__ float funflip(unsigned u) {
  u = (u & 0x80000000u) ? (u & 0x7fffffffu) : ~u;
  return __uint_as_float(u);
}

// ---- WMMA fragment loaders (16x16x32 bf16) --------------------------------
// A (16x32, MxK): lanes 0-15 row=lane, K = {0..7,16..23}; lanes 16-31 same
// rows, K shifted by 8 ({8..15,24..31}).
static __device__ __forceinline__ v16bf load_a_global_vec(const float* __restrict__ A,
                                                          long rowBase, int kb, int lane) {
  int kho = (lane & 16) ? 8 : 0;
  const float4* p = (const float4*)(A + rowBase + kb + kho);
  float4 a0 = p[0], a1 = p[1];            // K offsets 0..7
  float4 a2 = p[4], a3 = p[5];            // K offsets 16..23
  v16bf a;
  a[0] = f2bf(a0.x); a[1] = f2bf(a0.y); a[2]  = f2bf(a0.z); a[3]  = f2bf(a0.w);
  a[4] = f2bf(a1.x); a[5] = f2bf(a1.y); a[6]  = f2bf(a1.z); a[7]  = f2bf(a1.w);
  a[8] = f2bf(a2.x); a[9] = f2bf(a2.y); a[10] = f2bf(a2.z); a[11] = f2bf(a2.w);
  a[12] = f2bf(a3.x); a[13] = f2bf(a3.y); a[14] = f2bf(a3.z); a[15] = f2bf(a3.w);
  return a;
}
static __device__ __forceinline__ v16bf load_a_lds(const __bf16* S, int ld, int kb, int lane) {
  v16bf a;
  int row = lane & 15;
  int kho = (lane & 16) ? 8 : 0;
  const __bf16* p = S + row * ld + kb + kho;
#pragma unroll
  for (int j = 0; j < 8; ++j) { a[j] = p[j]; a[8 + j] = p[16 + j]; }
  return a;
}
// Fragment-order staging helper: element B[kk, col] of a KxN row-major matrix
// goes to slot [((tn*ksteps+ks)*32 + l)*16 + j] (l = lc + 16*((kp>>3)&1),
// j = (kp&7) + 8*(kp>>4)) so a B fragment is one contiguous 32-byte ds_load per lane.
static __device__ __forceinline__ int bfrag_slot(int kk, int col, int ksteps) {
  int tn = col >> 4, lc = col & 15;
  int kp = kk & 31, ks = kk >> 5;
  return ((tn * ksteps + ks) * 32 + lc + ((kp & 8) << 1)) * 16 + (kp & 7) + ((kp >> 1) & 8);
}

// ---- GEMM (N=128 fixed): C[Mx128] (+)= A[MxK] @ B[Kx128] + bias -----------
// Block = 8 waves; each wave owns one 16x128 strip (8 WMMA tiles, 64 acc
// VGPRs/lane). B is staged ONCE per block into LDS in fragment order, so each
// B-fragment read is a single 32-byte ds_load and each A-fragment (float4
// global_load_b128 x4) is reused by 8 WMMAs.
template <bool ACC>
__global__ __launch_bounds__(256) void gemm_n128_wmma_kernel(
    const float* __restrict__ A, const float* __restrict__ B,
    const float* __restrict__ bias, float* __restrict__ C, int M, int K) {
  constexpr int N = 128;
  __shared__ __bf16 sB[8 * 4 * 32 * 16];  // [tn][kstep][lane][j], 32 KB (K<=128)
  int tid = threadIdx.x;
  int lane = tid & 31;
  int wave = tid >> 5;
  int ksteps = K >> 5;

  // cooperative fragment-order staging of B (coalesced global reads)
  for (int idx = tid; idx < K * N; idx += 256) {
    int col = idx & 127, kk = idx >> 7;
    sB[bfrag_slot(kk, col, ksteps)] = f2bf(B[idx]);
  }
  __syncthreads();

  int tm = blockIdx.x * 8 + wave;
  if (tm >= (M >> 4)) return;  // wave-uniform: EXEC all-ones at every WMMA

  long rowBase = (long)(tm * 16 + (lane & 15)) * K;
  v8f acc[8] = {};
  for (int ks = 0; ks < ksteps; ++ks) {
    int kb = ks << 5;
    if (ks + 1 < ksteps)  // global_prefetch_b8 of next A k-slab
      __builtin_prefetch(A + rowBase + kb + 32, 0, 1);
    v16bf a = load_a_global_vec(A, rowBase, kb, lane);
#pragma unroll
    for (int tn = 0; tn < 8; ++tn) {
      v16bf b = *(const v16bf*)&sB[((tn * ksteps + ks) * 32 + lane) * 16];
      acc[tn] = __builtin_amdgcn_wmma_f32_16x16x32_bf16(false, a, false, b, (short)0,
                                                        acc[tn], false, false);
    }
  }

  int mbase = tm * 16 + ((lane >> 4) << 3);  // C: lanes 0-15 rows 0-7, 16-31 rows 8-15
#pragma unroll
  for (int tn = 0; tn < 8; ++tn) {
    int col = tn * 16 + (lane & 15);
    float bb = bias[col];
#pragma unroll
    for (int r = 0; r < 8; ++r) {
      long idx = (long)(mbase + r) * N + col;
      float val = acc[tn][r] + bb;
      if (ACC) val += C[idx];
      C[idx] = val;
    }
  }
}

// ---- fused neighbor MHA: 8 nodes per block (one wave32 per node) ----------
// Both weight matrices are staged once per block into fragment-order LDS
// (24 KB + 8 KB), cutting weight re-reads 8x vs one block per node.
// Per node: 10x64 -> qkv(16x192 WMMA) -> 4-head softmax(10x10) ->
// out-proj(16x64 WMMA) -> mean over 10 rows -> neim[node][64].
__global__ __launch_bounds__(256) void mha_fused_kernel(
    const float* __restrict__ nei,
    const float* __restrict__ in_w, const float* __restrict__ in_b,
    const float* __restrict__ out_w, const float* __restrict__ out_b,
    float* __restrict__ neim) {
  __shared__ __bf16 sWin[12 * 2 * 32 * 16];          // in_w fragments, 24 KB
  __shared__ __bf16 sWout[4 * 2 * 32 * 16];          // out_w fragments, 8 KB
  __shared__ __bf16 sA[MHA_WAVES][16 * 64];          // nei tiles, 16 KB
  __shared__ __bf16 sQKV[MHA_WAVES][16 * 192];       // qkv (bf16), 48 KB
  __shared__ __bf16 sO[MHA_WAVES][16 * 64];          // attn out, 16 KB
  int tid = threadIdx.x, lane = tid & 31, wave = tid >> 5;

  for (int idx = tid; idx < 64 * 192; idx += 256) {  // stage in_w (64x192)
    int col = idx % 192, kk = idx / 192;
    sWin[bfrag_slot(kk, col, 2)] = f2bf(in_w[idx]);
  }
  for (int idx = tid; idx < 64 * 64; idx += 256) {   // stage out_w (64x64)
    int col = idx & 63, kk = idx >> 6;
    sWout[bfrag_slot(kk, col, 2)] = f2bf(out_w[idx]);
  }
  __syncthreads();

  int node = blockIdx.x * MHA_WAVES + wave;
  if (node >= N_NODES) return;  // wave-uniform (30000 % 8 == 0 anyway)
  __bf16* mA = sA[wave];
  __bf16* mQKV = sQKV[wave];
  __bf16* mO = sO[wave];

  // load node's 10x64 tile (pad rows 10..15 with zeros), fp32 -> bf16
  const float* np = nei + (long)node * S_NEIC * D_NEIC;
  for (int idx = lane; idx < 16 * 64; idx += 32) {
    int r = idx >> 6, c = idx & 63;
    mA[idx] = f2bf(r < S_NEIC ? np[r * 64 + c] : 0.f);
    mO[idx] = f2bf(0.f);
  }

  // QKV = mA(16x64) @ in_w(64x192) + in_b  (bf16 result in LDS)
#pragma unroll
  for (int tn = 0; tn < 12; ++tn) {
    v8f acc = {};
#pragma unroll
    for (int ks = 0; ks < 2; ++ks) {
      v16bf a = load_a_lds(mA, 64, ks * 32, lane);
      v16bf b = *(const v16bf*)&sWin[((tn * 2 + ks) * 32 + lane) * 16];
      acc = __builtin_amdgcn_wmma_f32_16x16x32_bf16(false, a, false, b, (short)0, acc, false, false);
    }
    int col = tn * 16 + (lane & 15);
    float bb = in_b[col];
    int mbase = (lane >> 4) << 3;
#pragma unroll
    for (int r = 0; r < 8; ++r) mQKV[(mbase + r) * 192 + col] = f2bf(acc[r] + bb);
  }

  // per (head,row) softmax attention: hd=16, scale=1/sqrt(16)=0.25
  for (int w = lane; w < N_HEADSC * S_NEIC; w += 32) {
    int hh = w / S_NEIC, i = w % S_NEIC;
    float sc[S_NEIC];
    float mx = -1e30f;
#pragma unroll
    for (int j = 0; j < S_NEIC; ++j) {
      float d = 0.f;
#pragma unroll
      for (int c = 0; c < 16; ++c)
        d += (float)mQKV[i * 192 + hh * 16 + c] * (float)mQKV[j * 192 + 64 + hh * 16 + c];
      sc[j] = d * 0.25f;
      mx = fmaxf(mx, sc[j]);
    }
    float den = 0.f;
#pragma unroll
    for (int j = 0; j < S_NEIC; ++j) { sc[j] = __expf(sc[j] - mx); den += sc[j]; }
    float inv = 1.f / den;
#pragma unroll
    for (int c = 0; c < 16; ++c) {
      float o = 0.f;
#pragma unroll
      for (int j = 0; j < S_NEIC; ++j) o += sc[j] * (float)mQKV[j * 192 + 128 + hh * 16 + c];
      mO[i * 64 + hh * 16 + c] = f2bf(o * inv);
    }
  }

  // out = mO(16x64) @ out_w(64x64) + out_b, then mean over rows 0..9
#pragma unroll
  for (int tn = 0; tn < 4; ++tn) {
    v8f acc = {};
#pragma unroll
    for (int ks = 0; ks < 2; ++ks) {
      v16bf a = load_a_lds(mO, 64, ks * 32, lane);
      v16bf b = *(const v16bf*)&sWout[((tn * 2 + ks) * 32 + lane) * 16];
      acc = __builtin_amdgcn_wmma_f32_16x16x32_bf16(false, a, false, b, (short)0, acc, false, false);
    }
    int col = tn * 16 + (lane & 15);
    float bb = out_b[col];
    float s = 0.f;
    if (lane < 16) {  // rows 0..7
#pragma unroll
      for (int r = 0; r < 8; ++r) s += acc[r] + bb;
    } else {          // rows 8,9 only (10..15 are pad)
      s = (acc[0] + bb) + (acc[1] + bb);
    }
    s += __shfl_xor(s, 16, 32);
    if (lane < 16) neim[(long)node * 64 + col] = s * 0.1f;  // /S_NEI
  }
}

// ---- graph-conv edge pass 1: scores + segment max -------------------------
__global__ void edge_score_kernel(const float* __restrict__ q, const float* __restrict__ k,
                                  const int* __restrict__ src, const int* __restrict__ dst,
                                  float* __restrict__ score, unsigned* __restrict__ smax) {
  int t = blockIdx.x * blockDim.x + threadIdx.x;
  if (t >= N_EDGES * N_HEADSC) return;
  int e = t >> 2, hh = t & 3;
  int sn = src[e], dn = dst[e];
  const float4* qp = (const float4*)(q + (long)dn * D_HIDC + hh * 32);
  const float4* kp = (const float4*)(k + (long)sn * D_HIDC + hh * 32);
  float d = 0.f;
#pragma unroll
  for (int c = 0; c < 8; ++c) {
    float4 qv = qp[c], kv = kp[c];
    d += qv.x * kv.x + qv.y * kv.y + qv.z * kv.z + qv.w * kv.w;
  }
  d *= 0.17677669529663687f;  // 1/sqrt(32)
  score[t] = d;
  atomicMax(&smax[dn * 4 + hh], fflip(d));
}

// ---- graph-conv edge pass 2: exp + scatter-add num/denom ------------------
__global__ void edge_scatter_kernel(const float* __restrict__ v, const float* __restrict__ score,
                                    const unsigned* __restrict__ smax,
                                    const int* __restrict__ src, const int* __restrict__ dst,
                                    float* __restrict__ num, float* __restrict__ denom) {
  long t = (long)blockIdx.x * blockDim.x + threadIdx.x;
  if (t >= (long)N_EDGES * D_HIDC) return;
  int e = (int)(t >> 7), c = (int)(t & 127), hh = c >> 5;
  int sn = src[e], dn = dst[e];
  float ex = __expf(score[e * 4 + hh] - funflip(smax[dn * 4 + hh]));
  atomicAdd(&num[(long)dn * D_HIDC + c], ex * v[(long)sn * D_HIDC + c]);
  if ((c & 31) == 0) atomicAdd(&denom[dn * 4 + hh], ex);
}

// ---- out = LN(relu(num/denom)) , one wave32 per node ----------------------
__global__ void node_norm_kernel(const float* __restrict__ num, const float* __restrict__ denom,
                                 const float* __restrict__ g, const float* __restrict__ b,
                                 float* __restrict__ h) {
  int wave = threadIdx.x >> 5, lane = threadIdx.x & 31;
  int n = blockIdx.x * (blockDim.x >> 5) + wave;
  if (n >= N_NODES) return;
  float vals[4], s = 0.f, s2 = 0.f;
#pragma unroll
  for (int i = 0; i < 4; ++i) {
    int c = lane + 32 * i;
    float d = denom[n * 4 + i];
    float val = num[(long)n * D_HIDC + c] / fmaxf(d, 1e-9f);
    val = fmaxf(val, 0.f);  // relu
    vals[i] = val; s += val; s2 += val * val;
  }
#pragma unroll
  for (int off = 16; off; off >>= 1) { s += __shfl_xor(s, off, 32); s2 += __shfl_xor(s2, off, 32); }
  float mean = s * (1.f / 128.f);
  float var = s2 * (1.f / 128.f) - mean * mean;
  float rs = rsqrtf(var + 1e-5f);
#pragma unroll
  for (int i = 0; i < 4; ++i) {
    int c = lane + 32 * i;
    h[(long)n * D_HIDC + c] = (vals[i] - mean) * rs * g[c] + b[c];
  }
}

__global__ void logits_kernel(const float* __restrict__ h, const float* __restrict__ w_out,
                              const float* __restrict__ b_out, float* __restrict__ out) {
  int n = blockIdx.x * blockDim.x + threadIdx.x;
  if (n >= N_NODES) return;
  float s0 = b_out[0], s1 = b_out[1];
#pragma unroll 4
  for (int c = 0; c < D_HIDC; ++c) {
    float hv = h[(long)n * D_HIDC + c];
    s0 += hv * w_out[c * 2 + 0];
    s1 += hv * w_out[c * 2 + 1];
  }
  out[n * 2 + 0] = s0;
  out[n * 2 + 1] = s1;
}

__global__ void fill_zero_kernel(unsigned* __restrict__ p, long n) {
  long t = (long)blockIdx.x * blockDim.x + threadIdx.x;
  for (; t < n; t += (long)gridDim.x * blockDim.x) p[t] = 0u;
}

extern "C" void kernel_launch(void* const* d_in, const int* in_sizes, int n_in,
                              void* d_out, int out_size, void* d_ws, size_t ws_size,
                              hipStream_t stream) {
  const float* x         = (const float*)d_in[0];
  const float* nei       = (const float*)d_in[1];
  const int*   ei        = (const int*)d_in[2];
  const float* w_in      = (const float*)d_in[3];
  const float* b_in      = (const float*)d_in[4];
  const float* mha_in_w  = (const float*)d_in[5];
  const float* mha_in_b  = (const float*)d_in[6];
  const float* mha_out_w = (const float*)d_in[7];
  const float* mha_out_b = (const float*)d_in[8];
  const float* w_nei     = (const float*)d_in[9];
  const float* b_nei     = (const float*)d_in[10];
  const float* wq        = (const float*)d_in[11];
  const float* bq        = (const float*)d_in[12];
  const float* wk        = (const float*)d_in[13];
  const float* bk        = (const float*)d_in[14];
  const float* wv        = (const float*)d_in[15];
  const float* bv        = (const float*)d_in[16];
  const float* ln_g      = (const float*)d_in[17];
  const float* ln_b      = (const float*)d_in[18];
  const float* w_out     = (const float*)d_in[19];
  const float* b_out     = (const float*)d_in[20];
  float* out = (float*)d_out;

  // workspace carve-out (~93 MB, L2-resident working set)
  float* ws = (float*)d_ws;
  float* h     = ws; ws += (size_t)N_NODES * D_HIDC;
  float* neim  = ws; ws += (size_t)N_NODES * D_NEIC;
  float* q     = ws; ws += (size_t)N_NODES * D_HIDC;
  float* k     = ws; ws += (size_t)N_NODES * D_HIDC;
  float* v     = ws; ws += (size_t)N_NODES * D_HIDC;
  float* score = ws; ws += (size_t)N_EDGES * N_HEADSC;
  unsigned* smax = (unsigned*)ws; ws += (size_t)N_NODES * N_HEADSC;
  float* denom = ws; ws += (size_t)N_NODES * N_HEADSC;
  float* num   = ws; ws += (size_t)N_NODES * D_HIDC;

  const int BLK = 256, WAVES = BLK / 32;
  const int mTiles = N_NODES / 16;                     // 1875
  dim3 gGemm((mTiles + WAVES - 1) / WAVES);            // 235 blocks

  // h = x @ w_in + b_in
  gemm_n128_wmma_kernel<false><<<gGemm, BLK, 0, stream>>>(x, w_in, b_in, h, N_NODES, D_INC);
  // neighbor-MHA branch -> neim (8 nodes per block)
  mha_fused_kernel<<<N_NODES / MHA_WAVES, BLK, 0, stream>>>(nei, mha_in_w, mha_in_b,
                                                            mha_out_w, mha_out_b, neim);
  // h += neim @ w_nei + b_nei
  gemm_n128_wmma_kernel<true><<<gGemm, BLK, 0, stream>>>(neim, w_nei, b_nei, h, N_NODES, D_NEIC);

  for (int i = 0; i < N_LAYERS; ++i) {
    const int* src = ei + (size_t)i * 2 * N_EDGES;
    const int* dst = src + N_EDGES;
    gemm_n128_wmma_kernel<false><<<gGemm, BLK, 0, stream>>>(
        h, wq + (size_t)i * D_HIDC * D_HIDC, bq + (size_t)i * D_HIDC, q, N_NODES, D_HIDC);
    gemm_n128_wmma_kernel<false><<<gGemm, BLK, 0, stream>>>(
        h, wk + (size_t)i * D_HIDC * D_HIDC, bk + (size_t)i * D_HIDC, k, N_NODES, D_HIDC);
    gemm_n128_wmma_kernel<false><<<gGemm, BLK, 0, stream>>>(
        h, wv + (size_t)i * D_HIDC * D_HIDC, bv + (size_t)i * D_HIDC, v, N_NODES, D_HIDC);
    // zero smax+denom+num (contiguous)
    long zwords = (long)N_NODES * N_HEADSC * 2 + (long)N_NODES * D_HIDC;
    fill_zero_kernel<<<2048, BLK, 0, stream>>>(smax, zwords);
    edge_score_kernel<<<(N_EDGES * N_HEADSC + BLK - 1) / BLK, BLK, 0, stream>>>(q, k, src, dst, score, smax);
    edge_scatter_kernel<<<(int)(((long)N_EDGES * D_HIDC + BLK - 1) / BLK), BLK, 0, stream>>>(
        v, score, smax, src, dst, num, denom);
    node_norm_kernel<<<N_NODES / WAVES, BLK, 0, stream>>>(num, denom, ln_g + (size_t)i * D_HIDC,
                                                          ln_b + (size_t)i * D_HIDC, h);
  }
  logits_kernel<<<(N_NODES + BLK - 1) / BLK, BLK, 0, stream>>>(h, w_out, b_out, out);
}
